// HierarchicalLayer_88098369176163
// MI455X (gfx1250) — matrix operand order repaired and checked by reading, not compiled, not run on hardware
//
#include <hip/hip_runtime.h>

// ---- CDNA5 (gfx1250) types ------------------------------------------------
typedef _Float16 half_t;
typedef __attribute__((ext_vector_type(16))) _Float16 v16h;
typedef __attribute__((ext_vector_type(8)))  _Float16 v8h;
typedef __attribute__((ext_vector_type(8)))  float    v8f;

#define HID   64
#define HEADS 4
#define HB    8      // heads * 2 branches
#define PC    512    // HB * HID
#define SLOPE 0.2f

// ---- monotonic float<->uint encoding for atomic segment-max ---------------
__device__ __forceinline__ unsigned enc_f32(float f) {
  unsigned b = __float_as_uint(f);
  return (b & 0x80000000u) ? ~b : (b | 0x80000000u);
}
__device__ __forceinline__ float dec_f32(unsigned e) {
  unsigned b = (e & 0x80000000u) ? (e & 0x7fffffffu) : ~e;
  return __uint_as_float(b);
}

// ---- WMMA fragment loaders (contiguous k order, same for A and B) --------
// A (16x32 f16, row-major, row stride K): lane l holds row (l&15);
//   halves 0..15 = k in [off, off+15],
//   off = (l>=16)*16.  -> two contiguous 16B loads.
__device__ __forceinline__ v16h load_a_frag(const half_t* __restrict__ aBase, int k0) {
  v8h lo = *(const v8h*)(aBase + k0);
  v8h hi = *(const v8h*)(aBase + k0 + 8);
  v16h a;
#pragma unroll
  for (int i = 0; i < 8; ++i) { a[i] = lo[i]; a[i + 8] = hi[i]; }
  return a;
}
// B (32x16 f16) supplied pre-transposed Bt[Nc,K]: lane l holds col (l&15),
//   halves e = k0 + (l>=16)*16 + e, contiguous -> one 32B load.
__device__ __forceinline__ v16h load_b_frag(const half_t* __restrict__ bBase, int k0) {
  return *(const v16h*)(bBase + k0);
}

// ---- Plain WMMA GEMM: C[M,Nc] = A[M,K](f16,row-major) x Bt[Nc,K](f16) -----
// One 16x16 tile per wave; M, Nc, K multiples of 16/32. EXEC stays all-ones
// (tile-out-of-range exit is wave-uniform), as WMMA requires.
__global__ void wmma_gemm(const half_t* __restrict__ A, const half_t* __restrict__ Bt,
                          float* __restrict__ C, int M, int Nc, int K) {
  int wid  = blockIdx.x * (blockDim.x >> 5) + (threadIdx.x >> 5);
  int lane = threadIdx.x & 31;
  int ntn  = Nc >> 4;
  int total = (M >> 4) * ntn;
  if (wid >= total) return;
  int m0 = (wid / ntn) << 4;
  int n0 = (wid % ntn) << 4;

  const half_t* aBase = A  + (size_t)(m0 + (lane & 15)) * K + ((lane >> 4) << 4);
  const half_t* bBase = Bt + (size_t)(n0 + (lane & 15)) * K + ((lane >> 4) << 4);

  v8f c = {};
  for (int k0 = 0; k0 < K; k0 += 32) {
    v16h a = load_a_frag(aBase, k0);
    v16h b = load_b_frag(bBase, k0);
    c = __builtin_amdgcn_wmma_f32_16x16x32_f16(false, a, false, b, (short)0, c, false, false);
  }
  int cn    = n0 + (lane & 15);
  int mBase = m0 + ((lane >> 4) << 3);
#pragma unroll
  for (int j = 0; j < 8; ++j) C[(size_t)(mBase + j) * Nc + cn] = c[j];
}

// ---- WMMA GEMM + bias + ReLU epilogue; writes f16 next-x and f32 out slice
__global__ void wmma_gemm_lin(const half_t* __restrict__ A, const half_t* __restrict__ Bt,
                              const float* __restrict__ bias,
                              float* __restrict__ outp, int out_stride, int col0,
                              half_t* __restrict__ xh, int M, int Nc, int K) {
  int wid  = blockIdx.x * (blockDim.x >> 5) + (threadIdx.x >> 5);
  int lane = threadIdx.x & 31;
  int ntn  = Nc >> 4;
  int total = (M >> 4) * ntn;
  if (wid >= total) return;
  int m0 = (wid / ntn) << 4;
  int n0 = (wid % ntn) << 4;

  const half_t* aBase = A  + (size_t)(m0 + (lane & 15)) * K + ((lane >> 4) << 4);
  const half_t* bBase = Bt + (size_t)(n0 + (lane & 15)) * K + ((lane >> 4) << 4);

  v8f c = {};
  for (int k0 = 0; k0 < K; k0 += 32) {
    v16h a = load_a_frag(aBase, k0);
    v16h b = load_b_frag(bBase, k0);
    c = __builtin_amdgcn_wmma_f32_16x16x32_f16(false, a, false, b, (short)0, c, false, false);
  }
  int cn    = n0 + (lane & 15);
  int mBase = m0 + ((lane >> 4) << 3);
  float bv  = bias[cn];
#pragma unroll
  for (int j = 0; j < 8; ++j) {
    float v = fmaxf(c[j] + bv, 0.0f);
    int m = mBase + j;
    outp[(size_t)m * out_stride + col0 + cn] = v;
    xh[(size_t)m * Nc + cn] = (half_t)v;
  }
}

// ---- elementwise / setup kernels ------------------------------------------
__global__ void init_x_out(const float* __restrict__ h, half_t* __restrict__ xh,
                           float* __restrict__ outp, int N, int outc) {
  int i = blockIdx.x * blockDim.x + threadIdx.x;
  if (i >= N * HID) return;
  float v = h[i];
  xh[i] = (half_t)v;
  outp[(size_t)(i >> 6) * outc + (i & 63)] = v;
}

__global__ void f32_to_f16(const float* __restrict__ src, half_t* __restrict__ dst, int n) {
  int i = blockIdx.x * blockDim.x + threadIdx.x;
  if (i < n) dst[i] = (half_t)src[i];
}

// [W1|W2] (each [64,256], row-major) -> Wt[512,64] f16 (row n = column n of cat)
__global__ void conv_wp(const float* __restrict__ W1, const float* __restrict__ W2,
                        half_t* __restrict__ Wt) {
  int i = blockIdx.x * blockDim.x + threadIdx.x;
  if (i >= PC * HID) return;
  int n = i / HID, k = i % HID;
  float v = (n < 256) ? W1[(size_t)k * 256 + n] : W2[(size_t)k * 256 + (n - 256)];
  Wt[i] = (half_t)v;
}

// Wlin [512,64] row-major -> Wt[64,512] f16
__global__ void conv_wlin(const float* __restrict__ W, half_t* __restrict__ Wt) {
  int i = blockIdx.x * blockDim.x + threadIdx.x;
  if (i >= HID * PC) return;
  int n = i / PC, k = i % PC;
  Wt[i] = (half_t)W[(size_t)k * HID + n];
}

// es/ed[n,hb] = dot(P[n, hb*64 .. +63], a_{s,d}[branch][head])
__global__ void node_coeffs(const float* __restrict__ P,
                            const float* __restrict__ a1s, const float* __restrict__ a1d,
                            const float* __restrict__ a2s, const float* __restrict__ a2d,
                            float* __restrict__ es, float* __restrict__ ed, int N) {
  int idx = blockIdx.x * blockDim.x + threadIdx.x;
  if (idx >= N * HB) return;
  int n = idx >> 3, hb = idx & 7;
  const float* as = ((hb >> 2) ? a2s : a1s) + (hb & 3) * HID;
  const float* ad = ((hb >> 2) ? a2d : a1d) + (hb & 3) * HID;
  const float* p  = P + (size_t)n * PC + hb * HID;
  float s1 = 0.f, s2 = 0.f;
#pragma unroll 8
  for (int d = 0; d < HID; ++d) { s1 += p[d] * as[d]; s2 += p[d] * ad[d]; }
  es[idx] = s1; ed[idx] = s2;
}

__global__ void zero_init(float* __restrict__ O, unsigned* __restrict__ m_enc,
                          float* __restrict__ s, int totO, int nh) {
  int i = blockIdx.x * blockDim.x + threadIdx.x;
  if (i < totO) O[i] = 0.f;
  if (i < nh)  { m_enc[i] = 0u; s[i] = 0.f; }
}

// leaky-relu logit per (edge, head-branch); atomic segment-max into dst node
__global__ void edge_logit_max(const int* __restrict__ src, const int* __restrict__ dst,
                               const float* __restrict__ es, const float* __restrict__ ed,
                               float* __restrict__ ev, unsigned* __restrict__ m_enc,
                               int E, int Ef) {
  int t = blockIdx.x * blockDim.x + threadIdx.x;
  if (t >= Ef * HB) return;
  int e = t >> 3, hb = t & 7;
  int sn = (e < E) ? src[e] : (e - E);
  int dn = (e < E) ? dst[e] : (e - E);
  float l = es[sn * HB + hb] + ed[dn * HB + hb];
  l = (l > 0.f) ? l : SLOPE * l;
  ev[t] = l;
  atomicMax(&m_enc[dn * HB + hb], enc_f32(l));
}

__global__ void edge_exp_sum(const int* __restrict__ dst, float* __restrict__ ev,
                             const unsigned* __restrict__ m_enc, float* __restrict__ s,
                             int E, int Ef) {
  int t = blockIdx.x * blockDim.x + threadIdx.x;
  if (t >= Ef * HB) return;
  int e = t >> 3, hb = t & 7;
  int dn = (e < E) ? dst[e] : (e - E);
  float ex = __expf(ev[t] - dec_f32(m_enc[dn * HB + hb]));
  ev[t] = ex;
  atomicAdd(&s[dn * HB + hb], ex);
}

// O[dst, hb, :] += (ev/s[dst]) * P[src, hb, :]
__global__ void edge_agg(const int* __restrict__ src, const int* __restrict__ dst,
                         const float* __restrict__ ev, const float* __restrict__ s,
                         const float* __restrict__ P, float* __restrict__ O,
                         int E, int Ef) {
  int t = blockIdx.x * blockDim.x + threadIdx.x;
  if (t >= Ef * HB) return;
  int e = t >> 3, hb = t & 7;
  int sn = (e < E) ? src[e] : (e - E);
  int dn = (e < E) ? dst[e] : (e - E);
  float alpha = ev[t] / s[dn * HB + hb];
  const float4* p4 = (const float4*)(P + (size_t)sn * PC + hb * HID);
  float* o = O + (size_t)dn * PC + hb * HID;
#pragma unroll
  for (int q = 0; q < 16; ++q) {
    float4 v = p4[q];
    atomicAdd(o + q * 4 + 0, alpha * v.x);
    atomicAdd(o + q * 4 + 1, alpha * v.y);
    atomicAdd(o + q * 4 + 2, alpha * v.z);
    atomicAdd(o + q * 4 + 3, alpha * v.w);
  }
}

// ---- host driver -----------------------------------------------------------
static inline int cdiv(long a, long b) { return (int)((a + b - 1) / b); }

extern "C" void kernel_launch(void* const* d_in, const int* in_sizes, int n_in,
                              void* d_out, int out_size, void* d_ws, size_t ws_size,
                              hipStream_t stream) {
  const float* h    = (const float*)d_in[0];
  const int*   src  = (const int*)d_in[1];
  const int*   dst  = (const int*)d_in[2];
  const float* W1   = (const float*)d_in[3];
  const float* a1s  = (const float*)d_in[4];
  const float* a1d  = (const float*)d_in[5];
  const float* W2   = (const float*)d_in[6];
  const float* a2s  = (const float*)d_in[7];
  const float* a2d  = (const float*)d_in[8];
  const float* Wlin = (const float*)d_in[9];
  const float* blin = (const float*)d_in[10];
  float* outp = (float*)d_out;

  const int N  = in_sizes[0] / HID;   // 50000 (multiple of 16)
  const int E  = in_sizes[1];         // 800000
  const int L  = in_sizes[10] / HID;  // 5
  const int Ef = E + N;               // with self loops
  const int OUTC = (L + 1) * HID;     // 384

  // workspace carve-up (256B aligned)
  char* ws = (char*)d_ws; size_t off = 0;
  auto carve = [&](size_t bytes) -> void* {
    void* p = ws + off; off = (off + bytes + 255) & ~(size_t)255; return p;
  };
  half_t*   xh    = (half_t*)  carve((size_t)N * HID * 2);
  float*    P     = (float*)   carve((size_t)N * PC * 4);
  float*    O     = (float*)   carve((size_t)N * PC * 4);
  half_t*   Oh    = (half_t*)  carve((size_t)N * PC * 2);
  float*    es    = (float*)   carve((size_t)N * HB * 4);
  float*    ed    = (float*)   carve((size_t)N * HB * 4);
  unsigned* m_enc = (unsigned*)carve((size_t)N * HB * 4);
  float*    sseg  = (float*)   carve((size_t)N * HB * 4);
  float*    ev    = (float*)   carve((size_t)Ef * HB * 4);
  half_t*   WtP   = (half_t*)  carve((size_t)PC * HID * 2);
  half_t*   WtL   = (half_t*)  carve((size_t)HID * PC * 2);
  (void)ws_size; (void)n_in; (void)out_size;

  const int BLK = 256;

  // x0 = h : f16 copy + first output slice
  init_x_out<<<cdiv((long)N * HID, BLK), BLK, 0, stream>>>(h, xh, outp, N, OUTC);

  for (int i = 0; i < L; ++i) {
    const float* W1i = W1 + (size_t)i * HID * 256;
    const float* W2i = W2 + (size_t)i * HID * 256;
    const float* a1si = a1s + (size_t)i * HEADS * HID;
    const float* a1di = a1d + (size_t)i * HEADS * HID;
    const float* a2si = a2s + (size_t)i * HEADS * HID;
    const float* a2di = a2d + (size_t)i * HEADS * HID;
    const float* Wli = Wlin + (size_t)i * PC * HID;
    const float* bli = blin + (size_t)i * HID;

    // weight conversion / transposition to f16
    conv_wp  <<<cdiv(PC * HID, BLK), BLK, 0, stream>>>(W1i, W2i, WtP);
    conv_wlin<<<cdiv(HID * PC, BLK), BLK, 0, stream>>>(Wli, WtL);

    // P = x @ [W1|W2]  -> [N, 512]   (v_wmma_f32_16x16x32_f16)
    {
      int tiles = (N >> 4) * (PC >> 4);
      wmma_gemm<<<cdiv(tiles, BLK / 32), BLK, 0, stream>>>(xh, WtP, P, N, PC, HID);
    }

    // per-node attention coefficients
    node_coeffs<<<cdiv((long)N * HB, BLK), BLK, 0, stream>>>(P, a1si, a1di, a2si, a2di, es, ed, N);

    // init segment-max / segment-sum / output accumulator
    zero_init<<<cdiv((long)N * PC, BLK), BLK, 0, stream>>>(O, m_enc, sseg, N * PC, N * HB);

    // edge softmax (3 passes) + aggregation
    int eThreads = Ef * HB;
    edge_logit_max<<<cdiv(eThreads, BLK), BLK, 0, stream>>>(src, dst, es, ed, ev, m_enc, E, Ef);
    edge_exp_sum  <<<cdiv(eThreads, BLK), BLK, 0, stream>>>(dst, ev, m_enc, sseg, E, Ef);
    edge_agg      <<<cdiv(eThreads, BLK), BLK, 0, stream>>>(src, dst, ev, sseg, P, O, E, Ef);

    // x_{i+1} = relu(O @ Wlin + b); fused write of out slice + next f16 x
    f32_to_f16<<<cdiv((long)N * PC, BLK), BLK, 0, stream>>>(O, Oh, N * PC);
    {
      int tiles = (N >> 4) * (HID >> 4);
      wmma_gemm_lin<<<cdiv(tiles, BLK / 32), BLK, 0, stream>>>(
          Oh, WtL, bli, outp, OUTC, (i + 1) * HID, xh, N, HID, PC);
    }
  }
}
